// Encoder1_5566277616137
// MI455X (gfx1250) — compile-verified, matrix operand order checked by reading, not run on hardware
//
#include <hip/hip_runtime.h>
#include <hip/hip_bf16.h>
#include <stdint.h>

// ---------------------------------------------------------------------------
// 3-layer GCN (dense adjacency) for MI455X / gfx1250.
// - All big matmuls on v_wmma_f32_16x16x32_bf16 (wave32 WMMA, fixed VGPR acc).
// - hp and W stored PRE-TRANSPOSED so every LDS tile fill is contiguous b128
//   copies (round 2 showed scalar global_load_u16/ds_store_b16 gathers).
// - Aggregation GEMM (128 MB adjacency stream per layer) uses double-buffered
//   GLOBAL_LOAD_ASYNC_TO_LDS_B128 (ASYNCcnt) via inline asm to overlap the
//   HBM stream with WMMA.
// ---------------------------------------------------------------------------

#define NN 8192
#define NE 262144
#define DI 128
#define DH 64
#define DO 32
#define LDSTR 40   // ushort stride for LDS tiles: 80B rows -> 16B aligned

typedef __attribute__((ext_vector_type(16))) __bf16 v16bf;
typedef __attribute__((ext_vector_type(8)))  float  v8f;

union Frag { uint4 q[2]; v16bf v; };

__device__ __forceinline__ unsigned short f2bf(float f) {
  uint32_t u = __float_as_uint(f);
  u += 0x7FFFu + ((u >> 16) & 1u);          // round-to-nearest-even
  return (unsigned short)(u >> 16);
}
__device__ __forceinline__ float bf2f(unsigned short h) {
  return __uint_as_float(((uint32_t)h) << 16);
}
// counter-based uniform in [0,1): murmur-style 64-bit finalizer
__device__ __forceinline__ float u01(uint32_t a, uint32_t b) {
  uint64_t x = (((uint64_t)a) << 32) ^ (uint64_t)b;
  x ^= x >> 33; x *= 0xff51afd7ed558ccdULL;
  x ^= x >> 33; x *= 0xc4ceb9fe1a85ec53ULL;
  x ^= x >> 33;
  return (float)((uint32_t)x) * 2.3283064365386963e-10f;
}

// 16 bf16 (32 B, 2 x b128) global -> LDS, asynchronous (ASYNCcnt).
// INST_OFFSET is added to BOTH the LDS and global address (ISA 08 4.4),
// so one operand pair covers both 16B halves.
__device__ __forceinline__ void async_cp32(unsigned short* lds,
                                           const unsigned short* g) {
  uint32_t la = (uint32_t)(uintptr_t)lds;            // low 32b = LDS offset
  unsigned long long ga = (unsigned long long)(uintptr_t)g;
  asm volatile("global_load_async_to_lds_b128 %0, %1, off"
               :: "v"(la), "v"(ga) : "memory");
  asm volatile("global_load_async_to_lds_b128 %0, %1, off offset:16"
               :: "v"(la), "v"(ga) : "memory");
}
__device__ __forceinline__ void wait_async0() {
  asm volatile("s_wait_asynccnt 0x0" ::: "memory");
}

// ---------------- utility kernels ----------------
__global__ void zero_b128_kernel(uint4* p, size_t n16) {
  size_t i = (size_t)blockIdx.x * blockDim.x + threadIdx.x;
  size_t st = (size_t)gridDim.x * blockDim.x;
  uint4 z; z.x = z.y = z.z = z.w = 0u;
  for (; i < n16; i += st) p[i] = z;
}

__global__ void scatter_edges_kernel(const int* ei, unsigned short* adj0) {
  int e = blockIdx.x * blockDim.x + threadIdx.x;
  if (e >= NE) return;
  int r = ei[e];
  int c = ei[NE + e];
  adj0[(size_t)r * NN + c] = 0x3F80;  // bf16 1.0
}

__global__ void f32_to_bf16_kernel(const float* in, unsigned short* out, int n) {
  int i = blockIdx.x * blockDim.x + threadIdx.x;
  if (i < n) out[i] = f2bf(in[i]);
}

// Wt[n][k] = bf16(W[k][n])  (pre-transpose so GEMM B-tiles are contiguous)
__global__ void transpose_to_bf16_kernel(const float* W, unsigned short* Wt,
                                         int K, int M) {
  int i = blockIdx.x * blockDim.x + threadIdx.x;
  if (i < K * M) {
    int k = i / M, n = i % M;
    Wt[(size_t)n * K + k] = f2bf(W[i]);
  }
}

// dinv[row] = rsqrt(1 + sum_j A[row,j])   (self-loop adds the 1)
__global__ void row_dinv_kernel(const unsigned short* adj, float* dinv) {
  __shared__ float red[256];
  int row = blockIdx.x;
  const unsigned short* p = adj + (size_t)row * NN;
  float s = 0.f;
  for (int c = threadIdx.x; c < NN; c += 256) s += bf2f(p[c]);
  red[threadIdx.x] = s;
  __syncthreads();
  for (int off = 128; off; off >>= 1) {
    if (threadIdx.x < off) red[threadIdx.x] += red[threadIdx.x + off];
    __syncthreads();
  }
  if (threadIdx.x == 0) dinv[row] = rsqrtf(1.0f + red[0]);
}

// hpT[c][j] = bf16(dinv[j] * h[j][c])   (transposed for contiguous B-tiles)
__global__ void scale_rows_T_kernel(const float* h, const float* dinv,
                                    unsigned short* hpT, int M) {
  int i = blockIdx.x * blockDim.x + threadIdx.x;
  if (i < NN * M) {
    int j = i / M, c = i % M;
    hpT[(size_t)c * NN + j] = f2bf(h[i] * dinv[j]);
  }
}

// ---------------- WMMA GEMM: H = Zb @ W  (N x K) x (K x M), fp32 out -------
// WbT is the transposed weight [M][K].
template <int K, int M>
__global__ void gemm_xw_kernel(const unsigned short* __restrict__ Zb,
                               const unsigned short* __restrict__ WbT,
                               float* __restrict__ H) {
  constexpr int NT = M >> 4;
  __shared__ __align__(16) unsigned short As[64 * LDSTR];
  __shared__ __align__(16) unsigned short Bs[64 * LDSTR];
  const int tid = threadIdx.x, lane = tid & 31, wave = tid >> 5;
  const int rowBase = blockIdx.x * 64;
  const int r = tid >> 1, ch = (tid & 1) * 16;
  v8f acc[NT] = {};
#pragma unroll 1
  for (int k0 = 0; k0 < K; k0 += 32) {
    { // A tile 64x32, contiguous rows of Zb
      const unsigned short* src = Zb + (size_t)(rowBase + r) * K + k0 + ch;
      if (k0 + 32 < K) __builtin_prefetch(src + 32, 0, 3);  // global_prefetch_b8
      uint4* dst = (uint4*)(As + r * LDSTR + ch);
      dst[0] = ((const uint4*)src)[0]; dst[1] = ((const uint4*)src)[1];
    }
    if (tid < 2 * M) { // Bt tile (M rows): contiguous rows of WbT
      const uint4* src = (const uint4*)(WbT + (size_t)r * K + k0 + ch);
      uint4* dst = (uint4*)(Bs + r * LDSTR + ch);
      dst[0] = src[0]; dst[1] = src[1];
    }
    __syncthreads();
    int m = lane & 15, kb = (lane >> 4) * 8;
    Frag fa;
    fa.q[0] = *(const uint4*)(As + (wave * 16 + m) * LDSTR + kb);
    fa.q[1] = *(const uint4*)(As + (wave * 16 + m) * LDSTR + kb + 16);
#pragma unroll
    for (int nt = 0; nt < NT; ++nt) {
      Frag fb;
      fb.q[0] = *(const uint4*)(Bs + (nt * 16 + m) * LDSTR + kb);
      fb.q[1] = *(const uint4*)(Bs + (nt * 16 + m) * LDSTR + kb + 16);
      acc[nt] = __builtin_amdgcn_wmma_f32_16x16x32_bf16(
          false, fa.v, false, fb.v, (short)0, acc[nt], false, false);
    }
    __syncthreads();
  }
  int n0 = lane & 15, mh = (lane >> 4) * 8;
#pragma unroll
  for (int nt = 0; nt < NT; ++nt)
#pragma unroll
    for (int rr = 0; rr < 8; ++rr)
      H[(size_t)(rowBase + wave * 16 + mh + rr) * M + nt * 16 + n0] = acc[nt][rr];
}

// ------ WMMA GEMM (async pipelined): z = relu(dinv*(A@hp + dinv*h) + b) ----
template <int M>
__global__ void gemm_adj_kernel(const unsigned short* __restrict__ A,
                                const unsigned short* __restrict__ hpT,
                                const float* __restrict__ h,
                                const float* __restrict__ dinv,
                                const float* __restrict__ bias,
                                float* __restrict__ outF,
                                unsigned short* __restrict__ outB,
                                uint32_t seed) {
  constexpr int NT = M >> 4;
  __shared__ __align__(16) unsigned short As[2][64 * LDSTR];
  __shared__ __align__(16) unsigned short Bs[2][64 * LDSTR];
  const int tid = threadIdx.x, lane = tid & 31, wave = tid >> 5;
  const int rowBase = blockIdx.x * 64;
  const int r = tid >> 1, ch = (tid & 1) * 16;
  v8f acc[NT] = {};

  // per-thread stream pointers (advance by k0)
  const unsigned short* aSrc = A + (size_t)(rowBase + r) * NN + ch;
  const unsigned short* bSrc = hpT + (size_t)r * NN + ch;   // rows of hpT (r < M)

  // prologue: async-stage tile 0
  async_cp32(&As[0][r * LDSTR + ch], aSrc);
  if (tid < 2 * M) async_cp32(&Bs[0][r * LDSTR + ch], bSrc);

#pragma unroll 1
  for (int k0 = 0; k0 < NN; k0 += 32) {
    const int cur = (k0 >> 5) & 1, nxt = cur ^ 1;
    wait_async0();      // my copies for tile k0 have landed in LDS
    __syncthreads();    // whole tile visible; prior reads of buf[nxt] done
    if (k0 + 32 < NN) { // async-stage tile k0+32 while we compute on tile k0
      async_cp32(&As[nxt][r * LDSTR + ch], aSrc + k0 + 32);
      if (tid < 2 * M) async_cp32(&Bs[nxt][r * LDSTR + ch], bSrc + k0 + 32);
    }
    int m = lane & 15, kb = (lane >> 4) * 8;
    Frag fa;
    fa.q[0] = *(const uint4*)(&As[cur][(wave * 16 + m) * LDSTR + kb]);
    fa.q[1] = *(const uint4*)(&As[cur][(wave * 16 + m) * LDSTR + kb + 16]);
#pragma unroll
    for (int nt = 0; nt < NT; ++nt) {
      Frag fb;
      fb.q[0] = *(const uint4*)(&Bs[cur][(nt * 16 + m) * LDSTR + kb]);
      fb.q[1] = *(const uint4*)(&Bs[cur][(nt * 16 + m) * LDSTR + kb + 16]);
      acc[nt] = __builtin_amdgcn_wmma_f32_16x16x32_bf16(
          false, fa.v, false, fb.v, (short)0, acc[nt], false, false);
    }
  }
  int n0 = lane & 15, mh = (lane >> 4) * 8;
#pragma unroll
  for (int nt = 0; nt < NT; ++nt)
#pragma unroll
    for (int rr = 0; rr < 8; ++rr) {
      int i = rowBase + wave * 16 + mh + rr;
      int c = nt * 16 + n0;
      float di = dinv[i];
      // self-loop term + outer dinv scale + bias
      float v = (acc[nt][rr] + di * h[(size_t)i * M + c]) * di + bias[c];
      v = fmaxf(v, 0.0f);                                   // ReLU
      float u = u01(seed, (uint32_t)(i * M + c));           // Dropout(0.1)
      v = (u < 0.9f) ? v * (1.0f / 0.9f) : 0.0f;
      if (outF) outF[(size_t)i * M + c] = v;
      if (outB) outB[(size_t)i * M + c] = f2bf(v);
    }
}

// -------- WMMA GEMM: adjC = adj0 * (U < sigmoid(z @ z^T)), fused mask ------
template <int KD>
__global__ void gemm_zzt_mask_kernel(const unsigned short* __restrict__ Zb,
                                     const unsigned short* __restrict__ adj0,
                                     unsigned short* __restrict__ adjC,
                                     uint32_t seed) {
  __shared__ __align__(16) unsigned short As[64 * LDSTR];
  __shared__ __align__(16) unsigned short Bs[64 * LDSTR];
  const int tid = threadIdx.x, lane = tid & 31, wave = tid >> 5;
  const int iBase = blockIdx.x * 64, jBase = blockIdx.y * 64;
  const int r = tid >> 1, ch = (tid & 1) * 16;
  v8f acc[4] = {};
#pragma unroll 1
  for (int k0 = 0; k0 < KD; k0 += 32) {
    { // A rows from z[iBase..], Bt rows from z[jBase..] (B = z^T -> Bt = z)
      const uint4* sa = (const uint4*)(Zb + (size_t)(iBase + r) * KD + k0 + ch);
      const uint4* sb = (const uint4*)(Zb + (size_t)(jBase + r) * KD + k0 + ch);
      uint4* da = (uint4*)(As + r * LDSTR + ch);
      uint4* db = (uint4*)(Bs + r * LDSTR + ch);
      da[0] = sa[0]; da[1] = sa[1];
      db[0] = sb[0]; db[1] = sb[1];
    }
    __syncthreads();
    int m = lane & 15, kb = (lane >> 4) * 8;
    Frag fa;
    fa.q[0] = *(const uint4*)(As + (wave * 16 + m) * LDSTR + kb);
    fa.q[1] = *(const uint4*)(As + (wave * 16 + m) * LDSTR + kb + 16);
#pragma unroll
    for (int nt = 0; nt < 4; ++nt) {
      Frag fb;
      fb.q[0] = *(const uint4*)(Bs + (nt * 16 + m) * LDSTR + kb);
      fb.q[1] = *(const uint4*)(Bs + (nt * 16 + m) * LDSTR + kb + 16);
      acc[nt] = __builtin_amdgcn_wmma_f32_16x16x32_bf16(
          false, fa.v, false, fb.v, (short)0, acc[nt], false, false);
    }
    __syncthreads();
  }
  int n0 = lane & 15, mh = (lane >> 4) * 8;
#pragma unroll
  for (int nt = 0; nt < 4; ++nt)
#pragma unroll
    for (int rr = 0; rr < 8; ++rr) {
      int i = iBase + wave * 16 + mh + rr;
      int j = jBase + nt * 16 + n0;
      size_t idx = (size_t)i * NN + j;
      float s = 1.0f / (1.0f + __expf(-acc[nt][rr]));       // sigmoid
      float u = u01(seed ^ (uint32_t)(idx >> 32), (uint32_t)idx);
      adjC[idx] = (u < s) ? adj0[idx] : (unsigned short)0;  // adj0 * sample
    }
}

// ---------------------------------------------------------------------------
extern "C" void kernel_launch(void* const* d_in, const int* in_sizes, int n_in,
                              void* d_out, int out_size, void* d_ws, size_t ws_size,
                              hipStream_t stream) {
  const float* x = (const float*)d_in[0];
  const float* W0 = (const float*)d_in[1];
  const float* b0 = (const float*)d_in[2];
  const float* W1 = (const float*)d_in[3];
  const float* b1 = (const float*)d_in[4];
  const float* W2 = (const float*)d_in[5];
  const float* b2 = (const float*)d_in[6];
  const int* ei = (const int*)d_in[7];
  float* out = (float*)d_out;
  (void)in_sizes; (void)n_in; (void)out_size; (void)ws_size;

  const size_t N2 = (size_t)NN * NN;
  unsigned short* adj0 = (unsigned short*)d_ws;            // 128 MB bf16
  unsigned short* adjC = adj0 + N2;                        // 128 MB bf16
  float*          h    = (float*)(adjC + N2);              // N x 64 fp32
  unsigned short* zb   = (unsigned short*)(h + (size_t)NN * 64);  // N x 128 bf16
  unsigned short* hpT  = zb + (size_t)NN * 128;            // 64 x N bf16 (transposed)
  float*          dinv = (float*)(hpT + (size_t)NN * 64);  // N fp32
  unsigned short* WbT  = (unsigned short*)(dinv + NN);     // 64 x 128 bf16 (transposed)

  // adjacency of original graph (binary, bf16), then working copy
  zero_b128_kernel<<<4096, 256, 0, stream>>>((uint4*)adj0, N2 * 2 / 16);
  scatter_edges_kernel<<<NE / 256, 256, 0, stream>>>(ei, adj0);
  hipMemcpyAsync(adjC, adj0, N2 * 2, hipMemcpyDeviceToDevice, stream);

  // x -> bf16
  f32_to_bf16_kernel<<<(NN * DI + 255) / 256, 256, 0, stream>>>(x, zb, NN * DI);

  // ---------------- layer 0: K=128 -> M=64 ----------------
  transpose_to_bf16_kernel<<<(DI * DH + 255) / 256, 256, 0, stream>>>(W0, WbT, DI, DH);
  gemm_xw_kernel<DI, DH><<<NN / 64, 128, 0, stream>>>(zb, WbT, h);
  row_dinv_kernel<<<NN, 256, 0, stream>>>(adjC, dinv);
  scale_rows_T_kernel<<<(NN * DH + 255) / 256, 256, 0, stream>>>(h, dinv, hpT, DH);
  gemm_adj_kernel<DH><<<NN / 64, 128, 0, stream>>>(
      adjC, hpT, h, dinv, b0, nullptr, zb, 0x9E3779B9u + 0);
  gemm_zzt_mask_kernel<DH><<<dim3(NN / 64, NN / 64), 128, 0, stream>>>(
      zb, adj0, adjC, 0xC2B2AE35u + 0);

  // ---------------- layer 1: K=64 -> M=64 ----------------
  transpose_to_bf16_kernel<<<(DH * DH + 255) / 256, 256, 0, stream>>>(W1, WbT, DH, DH);
  gemm_xw_kernel<DH, DH><<<NN / 64, 128, 0, stream>>>(zb, WbT, h);
  row_dinv_kernel<<<NN, 256, 0, stream>>>(adjC, dinv);
  scale_rows_T_kernel<<<(NN * DH + 255) / 256, 256, 0, stream>>>(h, dinv, hpT, DH);
  gemm_adj_kernel<DH><<<NN / 64, 128, 0, stream>>>(
      adjC, hpT, h, dinv, b1, nullptr, zb, 0x9E3779B9u + 1);
  gemm_zzt_mask_kernel<DH><<<dim3(NN / 64, NN / 64), 128, 0, stream>>>(
      zb, adj0, adjC, 0xC2B2AE35u + 1);

  // ---------------- layer 2: K=64 -> M=32 (final output) ----------------
  transpose_to_bf16_kernel<<<(DH * DO + 255) / 256, 256, 0, stream>>>(W2, WbT, DH, DO);
  gemm_xw_kernel<DH, DO><<<NN / 64, 128, 0, stream>>>(zb, WbT, h);
  row_dinv_kernel<<<NN, 256, 0, stream>>>(adjC, dinv);
  scale_rows_T_kernel<<<(NN * DO + 255) / 256, 256, 0, stream>>>(h, dinv, hpT, DO);
  gemm_adj_kernel<DO><<<NN / 64, 128, 0, stream>>>(
      adjC, hpT, h, dinv, b2, out, nullptr, 0x9E3779B9u + 2);
}
